// QuantileSAE_82480551952999
// MI455X (gfx1250) — matrix operand order repaired and checked by reading, not compiled, not run on hardware
//
#include <hip/hip_runtime.h>
#include <hip/hip_bf16.h>

typedef __attribute__((ext_vector_type(16))) __bf16 v16bf;
typedef __attribute__((ext_vector_type(8)))  float  v8f;

union FragBf { v16bf v; uint4 q[2]; };

constexpr int BATCH = 4096;
constexpr int ACT   = 768;
constexpr int DICT  = 12288;
constexpr int NDEAD = 1024;
constexpr int NBLKN = DICT / 128;   // 96 N-blocks in encoder GEMM

// ---------------- workspace layout (bytes) ----------------
constexpr size_t OFF_XN      = 0;
constexpr size_t SZ_XN       = (size_t)BATCH * ACT * 4;
constexpr size_t OFF_XSTD    = OFF_XN + SZ_XN;
constexpr size_t SZ_XSTD     = (size_t)BATCH * 4;
constexpr size_t OFF_XCENT   = OFF_XSTD + SZ_XSTD;
constexpr size_t SZ_XCENT    = (size_t)BATCH * ACT * 2;
constexpr size_t OFF_WENCT   = OFF_XCENT + SZ_XCENT;                 // [DICT][ACT] bf16
constexpr size_t SZ_WENCT    = (size_t)DICT * ACT * 2;
constexpr size_t OFF_WDECT   = OFF_WENCT + SZ_WENCT;                 // [ACT][DICT] bf16
constexpr size_t SZ_WDECT    = (size_t)ACT * DICT * 2;
constexpr size_t OFF_THRBF   = OFF_WDECT + SZ_WDECT;                 // [BATCH][DICT] bf16
constexpr size_t SZ_THRBF    = (size_t)BATCH * DICT * 2;
constexpr size_t OFF_PREDEAD = OFF_THRBF + SZ_THRBF;                 // [BATCH][NDEAD] f32
constexpr size_t SZ_PREDEAD  = (size_t)BATCH * NDEAD * 4;
constexpr size_t OFF_MB      = OFF_PREDEAD + SZ_PREDEAD;             // [BATCH][NBLKN] f32
constexpr size_t SZ_PART     = (size_t)BATCH * NBLKN * 4;
constexpr size_t OFF_SB      = OFF_MB  + SZ_PART;
constexpr size_t OFF_L1P     = OFF_SB  + SZ_PART;
constexpr size_t OFF_L0P     = OFF_L1P + SZ_PART;
constexpr size_t OFF_AUX     = OFF_L0P + SZ_PART;                    // [BATCH][NDEAD] bf16
constexpr size_t SZ_AUX      = (size_t)BATCH * NDEAD * 2;
constexpr size_t OFF_SOFTD   = OFF_AUX + SZ_AUX;                     // [BATCH][NDEAD] bf16
constexpr size_t OFF_XREC    = OFF_SOFTD + SZ_AUX;                   // [BATCH][ACT] f32
constexpr size_t OFF_RESID   = OFF_XREC + SZ_XN;
constexpr size_t OFF_ACC     = OFF_RESID + SZ_XN;                    // 8 floats

// output layout (floats)
constexpr size_t OUT_SAE  = 0;
constexpr size_t OUT_ACTS = (size_t)BATCH * ACT;
constexpr size_t OUT_SCAL = OUT_ACTS + (size_t)BATCH * DICT;

// order-preserving float<->uint map for atomic max
__device__ __forceinline__ unsigned f2ord(float x) {
  unsigned u = __float_as_uint(x);
  return u ^ ((u >> 31) ? 0xffffffffu : 0x80000000u);
}
__device__ __forceinline__ float ord2f(unsigned m) {
  return __uint_as_float(m ^ ((m >> 31) ? 0x80000000u : 0xffffffffu));
}

// async global->LDS copy of 32 bytes (two b128 beats), ASYNCcnt-tracked
__device__ __forceinline__ void asyncCopy32(unsigned ldsAddr, const __bf16* gptr) {
  asm volatile(
      "global_load_async_to_lds_b128 %0, %1, off\n\t"
      "global_load_async_to_lds_b128 %0, %1, off offset:16"
      :: "v"(ldsAddr), "v"(gptr) : "memory");
}
__device__ __forceinline__ void waitAsync0() {
  asm volatile("s_wait_asynccnt 0x0" ::: "memory");
}

__device__ __forceinline__ float blockReduceSum256(float v, float* sred) {
  __syncthreads();
  int t = threadIdx.x;
  sred[t] = v;
  __syncthreads();
  for (int s = 128; s > 0; s >>= 1) {
    if (t < s) sred[t] += sred[t + s];
    __syncthreads();
  }
  return sred[0];
}

// ---------------- init ----------------
__global__ void k_init(float* acc) {
  if (threadIdx.x < 8) acc[threadIdx.x] = 0.f;
}

// ---------------- row normalize ----------------
__global__ __launch_bounds__(256) void k_norm(const float* __restrict__ x,
                                              const float* __restrict__ bdec,
                                              float* __restrict__ xn,
                                              float* __restrict__ xstd,
                                              __bf16* __restrict__ xcent) {
  __shared__ float sred[256];
  const int row = blockIdx.x, t = threadIdx.x;
  float v[3]; float s = 0.f;
#pragma unroll
  for (int i = 0; i < 3; ++i) { v[i] = x[(size_t)row * ACT + t + i * 256]; s += v[i]; }
  float mean = blockReduceSum256(s, sred) * (1.f / (float)ACT);
  float s2 = 0.f;
#pragma unroll
  for (int i = 0; i < 3; ++i) { v[i] -= mean; s2 += v[i] * v[i]; }
  float var = blockReduceSum256(s2, sred) * (1.f / (float)(ACT - 1));
  float stdv = sqrtf(var);
  float inv = 1.f / (stdv + 1e-5f);
  if (t == 0) xstd[row] = stdv;
#pragma unroll
  for (int i = 0; i < 3; ++i) {
    int c = t + i * 256;
    float nx = v[i] * inv;
    xn[(size_t)row * ACT + c] = nx;
    xcent[(size_t)row * ACT + c] = (__bf16)(nx - bdec[c]);
  }
}

// ---------------- transpose f32 -> bf16, dst[C][R] = src[R][C] ----------------
__global__ __launch_bounds__(256) void k_transpose(const float* __restrict__ src,
                                                   __bf16* __restrict__ dst,
                                                   int R, int C) {
  __shared__ float tile[32][33];
  const int c0 = blockIdx.x * 32, r0 = blockIdx.y * 32;
  const int tx = threadIdx.x & 31, ty = threadIdx.x >> 5;
#pragma unroll
  for (int i = ty; i < 32; i += 8)
    tile[i][tx] = src[(size_t)(r0 + i) * C + c0 + tx];
  __syncthreads();
#pragma unroll
  for (int i = ty; i < 32; i += 8)
    dst[(size_t)(c0 + i) * R + r0 + tx] = (__bf16)tile[tx][i];
}

// ---------------- encoder GEMM: acts_pre = x_cent @ W_enc + b_enc ----------------
// double-buffered async-to-LDS pipeline
__global__ __launch_bounds__(256, 1) void k_gemm_enc(
    const __bf16* __restrict__ A,   // [BATCH][ACT]
    const __bf16* __restrict__ Bt,  // [DICT][ACT]
    const float* __restrict__ benc,
    const float* __restrict__ qptr,
    float*  __restrict__ thrOut,    // d_out acts region [BATCH][DICT]
    __bf16* __restrict__ thrBf,     // ws bf16 copy
    float*  __restrict__ preDead,   // [BATCH][NDEAD]
    float* __restrict__ Mb, float* __restrict__ Sb,
    float* __restrict__ L1p, float* __restrict__ L0p) {
  __shared__ __bf16 sA[2][128 * 40];
  __shared__ __bf16 sB[2][128 * 40];
  __shared__ float sbias[128];
  __shared__ unsigned rMax[128];
  __shared__ float rL1[128], rL0[128], rS[128];

  const int bn = blockIdx.x, bm = blockIdx.y;
  const int tid = threadIdx.x, lane = tid & 31, w = tid >> 5;
  const int wm = w >> 1, wn = w & 1;
  const int rowBase = bm * 128, colBase = bn * 128;

  if (tid < 128) {
    sbias[tid] = benc[colBase + tid];
    rMax[tid] = 0u; rL1[tid] = 0.f; rL0[tid] = 0.f; rS[tid] = 0.f;
  }

  const v8f vz = {0.f,0.f,0.f,0.f,0.f,0.f,0.f,0.f};
  v8f acc[2][4];
#pragma unroll
  for (int i = 0; i < 2; ++i)
#pragma unroll
    for (int j = 0; j < 4; ++j) acc[i][j] = vz;

  const int ldr = tid >> 1, ldh = (tid & 1) * 16;
  const int mlo = lane & 15, hsel = lane >> 4;
  const int kb = 8 * hsel, kh = 16 * hsel;

  const __bf16* gA = A  + (size_t)(rowBase + ldr) * ACT + ldh;
  const __bf16* gB = Bt + (size_t)(colBase + ldr) * ACT + ldh;
  const unsigned lA0 = (unsigned)(size_t)&sA[0][ldr * 40 + ldh];
  const unsigned lA1 = (unsigned)(size_t)&sA[1][ldr * 40 + ldh];
  const unsigned lB0 = (unsigned)(size_t)&sB[0][ldr * 40 + ldh];
  const unsigned lB1 = (unsigned)(size_t)&sB[1][ldr * 40 + ldh];

  // prologue: fill buffer 0
  asyncCopy32(lA0, gA);
  asyncCopy32(lB0, gB);

  for (int k0 = 0; k0 < ACT; k0 += 32) {
    const int cur = (k0 >> 5) & 1;
    waitAsync0();          // my async writes into buf[cur] have landed
    __syncthreads();       // everyone's landed; prev reads of buf[1-cur] drained
    if (k0 + 32 < ACT) {
      asyncCopy32(cur ? lA0 : lA1, gA + k0 + 32);
      asyncCopy32(cur ? lB0 : lB1, gB + k0 + 32);
      if (k0 + 64 < ACT) __builtin_prefetch(gA + k0 + 64, 0, 1);
    }

    FragBf aF[2], bF[4];
#pragma unroll
    for (int i = 0; i < 2; ++i) {
      int m = wm * 32 + i * 16 + mlo;
      aF[i].q[0] = *(const uint4*)&sA[cur][m * 40 + kb];
      aF[i].q[1] = *(const uint4*)&sA[cur][m * 40 + 16 + kb];
    }
#pragma unroll
    for (int j = 0; j < 4; ++j) {
      int n = wn * 64 + j * 16 + mlo;
      bF[j].q[0] = *(const uint4*)&sB[cur][n * 40 + kb];
      bF[j].q[1] = *(const uint4*)&sB[cur][n * 40 + 16 + kb];
    }
#pragma unroll
    for (int i = 0; i < 2; ++i)
#pragma unroll
      for (int j = 0; j < 4; ++j)
        acc[i][j] = __builtin_amdgcn_wmma_f32_16x16x32_bf16(
            false, aF[i].v, false, bF[j].v, (short)0, acc[i][j], false, false);
  }
  __syncthreads();

  const float q = *qptr;
#pragma unroll
  for (int i = 0; i < 2; ++i)
#pragma unroll
    for (int j = 0; j < 4; ++j)
#pragma unroll
      for (int r = 0; r < 8; ++r) {
        int mrow = wm * 32 + i * 16 + 8 * hsel + r;
        int ncol = wn * 64 + j * 16 + mlo;
        float p = acc[i][j][r] + sbias[ncol];
        acc[i][j][r] = p;
        float rel = fmaxf(p, 0.f);
        float thr = (rel > q) ? rel : 0.f;
        size_t gi = (size_t)(rowBase + mrow) * DICT + colBase + ncol;
        thrOut[gi] = thr;
        thrBf[gi]  = (__bf16)thr;
        int gcol = colBase + ncol;
        if (gcol < NDEAD) preDead[(size_t)(rowBase + mrow) * NDEAD + gcol] = p;
        atomicMax(&rMax[mrow], f2ord(p));
        if (thr > 0.f) { atomicAdd(&rL1[mrow], thr); atomicAdd(&rL0[mrow], 1.f); }
      }
  __syncthreads();
#pragma unroll
  for (int i = 0; i < 2; ++i)
#pragma unroll
    for (int j = 0; j < 4; ++j)
#pragma unroll
      for (int r = 0; r < 8; ++r) {
        int mrow = wm * 32 + i * 16 + 8 * hsel + r;
        float p = acc[i][j][r];
        atomicAdd(&rS[mrow], __expf(p - ord2f(rMax[mrow])));
      }
  __syncthreads();
  if (tid < 128) {
    size_t pi = (size_t)(rowBase + tid) * NBLKN + bn;
    Mb[pi]  = ord2f(rMax[tid]);
    Sb[pi]  = rS[tid];
    L1p[pi] = rL1[tid];
    L0p[pi] = rL0[tid];
  }
}

// ---------------- per-row: softmax stats combine, softmax-aux, top-512 ----------------
__global__ __launch_bounds__(512, 1) void k_rowaux(
    const float* __restrict__ preDead,
    const float* __restrict__ Mb, const float* __restrict__ Sb,
    const float* __restrict__ L1p, const float* __restrict__ L0p,
    __bf16* __restrict__ softdead, __bf16* __restrict__ auxOut,
    float* __restrict__ accg) {
  __shared__ float sv[1024];
  __shared__ int   si[1024];
  __shared__ float sax[1024];
  __shared__ float sr[128];
  __shared__ float shM, shS;
  const int row = blockIdx.x, t = threadIdx.x;

  // combine 96 per-block partials: max
  float mv = -3.4e38f;
  if (t < 96) mv = Mb[(size_t)row * NBLKN + t];
  if (t < 128) sr[t] = mv;
  __syncthreads();
  for (int s = 64; s > 0; s >>= 1) { if (t < s) sr[t] = fmaxf(sr[t], sr[t + s]); __syncthreads(); }
  if (t == 0) shM = sr[0];
  __syncthreads();
  const float M = shM;

  // sum-exp merge
  float sval = 0.f;
  if (t < 96) { size_t pi = (size_t)row * NBLKN + t; sval = Sb[pi] * __expf(Mb[pi] - M); }
  if (t < 128) sr[t] = sval;
  __syncthreads();
  for (int s = 64; s > 0; s >>= 1) { if (t < s) sr[t] += sr[t + s]; __syncthreads(); }
  if (t == 0) shS = sr[0];
  __syncthreads();
  const float S = shS;

  // l1 row sum
  float l1v = (t < 96) ? L1p[(size_t)row * NBLKN + t] : 0.f;
  if (t < 128) sr[t] = l1v;
  __syncthreads();
  for (int s = 64; s > 0; s >>= 1) { if (t < s) sr[t] += sr[t + s]; __syncthreads(); }
  if (t == 0) atomicAdd(&accg[3], sr[0]);
  __syncthreads();
  // l0 row sum
  float l0v = (t < 96) ? L0p[(size_t)row * NBLKN + t] : 0.f;
  if (t < 128) sr[t] = l0v;
  __syncthreads();
  for (int s = 64; s > 0; s >>= 1) { if (t < s) sr[t] += sr[t + s]; __syncthreads(); }
  if (t == 0) atomicAdd(&accg[4], sr[0]);

  // softmax-weighted dead pre-acts; stage relu acts for top-k
  const float invS = 64.f / S;
#pragma unroll
  for (int c = t; c < NDEAD; c += 512) {
    float p = preDead[(size_t)row * NDEAD + c];
    float a = fmaxf(p, 0.f);
    softdead[(size_t)row * NDEAD + c] = (__bf16)(p * __expf(p - M) * invS);
    sv[c] = a; si[c] = c;
  }
  __syncthreads();

  // bitonic sort 1024 descending (value, index payload)
  for (int k = 2; k <= 1024; k <<= 1)
    for (int j = k >> 1; j > 0; j >>= 1) {
      int a = ((t & ~(j - 1)) << 1) | (t & (j - 1));
      int b = a | j;
      bool asc = (a & k) != 0;
      float va = sv[a], vb = sv[b];
      bool sw = asc ? (va > vb) : (va < vb);
      if (sw) {
        sv[a] = vb; sv[b] = va;
        int ia = si[a]; si[a] = si[b]; si[b] = ia;
      }
      __syncthreads();
    }

#pragma unroll
  for (int c = t; c < NDEAD; c += 512) sax[c] = 0.f;
  __syncthreads();
  sax[si[t]] = sv[t];       // t < 512: winners
  __syncthreads();
#pragma unroll
  for (int c = t; c < NDEAD; c += 512)
    auxOut[(size_t)row * NDEAD + c] = (__bf16)sax[c];
}

// ---------------- decoder GEMM: x_rec = acts_thr @ W_dec + b_dec ----------------
// double-buffered async-to-LDS pipeline
__global__ __launch_bounds__(256, 1) void k_gemm_dec(
    const __bf16* __restrict__ A,   // [BATCH][DICT]
    const __bf16* __restrict__ Bt,  // [ACT][DICT]
    const float* __restrict__ bdec,
    float* __restrict__ xrec) {
  __shared__ __bf16 sA[2][128 * 40];
  __shared__ __bf16 sB[2][128 * 40];
  __shared__ float sbias[128];

  const int bn = blockIdx.x, bm = blockIdx.y;
  const int tid = threadIdx.x, lane = tid & 31, w = tid >> 5;
  const int wm = w >> 1, wn = w & 1;
  const int rowBase = bm * 128, colBase = bn * 128;
  if (tid < 128) sbias[tid] = bdec[colBase + tid];

  const v8f vz = {0.f,0.f,0.f,0.f,0.f,0.f,0.f,0.f};
  v8f acc[2][4];
#pragma unroll
  for (int i = 0; i < 2; ++i)
#pragma unroll
    for (int j = 0; j < 4; ++j) acc[i][j] = vz;

  const int ldr = tid >> 1, ldh = (tid & 1) * 16;
  const int mlo = lane & 15, hsel = lane >> 4;
  const int kb = 8 * hsel, kh = 16 * hsel;

  const __bf16* gA = A  + (size_t)(rowBase + ldr) * DICT + ldh;
  const __bf16* gB = Bt + (size_t)(colBase + ldr) * DICT + ldh;
  const unsigned lA0 = (unsigned)(size_t)&sA[0][ldr * 40 + ldh];
  const unsigned lA1 = (unsigned)(size_t)&sA[1][ldr * 40 + ldh];
  const unsigned lB0 = (unsigned)(size_t)&sB[0][ldr * 40 + ldh];
  const unsigned lB1 = (unsigned)(size_t)&sB[1][ldr * 40 + ldh];

  asyncCopy32(lA0, gA);
  asyncCopy32(lB0, gB);

  for (int k0 = 0; k0 < DICT; k0 += 32) {
    const int cur = (k0 >> 5) & 1;
    waitAsync0();
    __syncthreads();
    if (k0 + 32 < DICT) {
      asyncCopy32(cur ? lA0 : lA1, gA + k0 + 32);
      asyncCopy32(cur ? lB0 : lB1, gB + k0 + 32);
      if (k0 + 64 < DICT) {
        __builtin_prefetch(gA + k0 + 64, 0, 1);
        __builtin_prefetch(gB + k0 + 64, 0, 1);
      }
    }

    FragBf aF[2], bF[4];
#pragma unroll
    for (int i = 0; i < 2; ++i) {
      int m = wm * 32 + i * 16 + mlo;
      aF[i].q[0] = *(const uint4*)&sA[cur][m * 40 + kb];
      aF[i].q[1] = *(const uint4*)&sA[cur][m * 40 + 16 + kb];
    }
#pragma unroll
    for (int j = 0; j < 4; ++j) {
      int n = wn * 64 + j * 16 + mlo;
      bF[j].q[0] = *(const uint4*)&sB[cur][n * 40 + kb];
      bF[j].q[1] = *(const uint4*)&sB[cur][n * 40 + 16 + kb];
    }
#pragma unroll
    for (int i = 0; i < 2; ++i)
#pragma unroll
      for (int j = 0; j < 4; ++j)
        acc[i][j] = __builtin_amdgcn_wmma_f32_16x16x32_bf16(
            false, aF[i].v, false, bF[j].v, (short)0, acc[i][j], false, false);
  }
  __syncthreads();
#pragma unroll
  for (int i = 0; i < 2; ++i)
#pragma unroll
    for (int j = 0; j < 4; ++j)
#pragma unroll
      for (int r = 0; r < 8; ++r) {
        int mrow = wm * 32 + i * 16 + 8 * hsel + r;
        int ncol = wn * 64 + j * 16 + mlo;
        xrec[(size_t)(rowBase + mrow) * ACT + colBase + ncol] = acc[i][j][r] + sbias[ncol];
      }
}

// ---------------- residual / sae_out / l2 ----------------
__global__ __launch_bounds__(256) void k_resid(const float* __restrict__ xn,
                                               const float* __restrict__ xstd,
                                               const float* __restrict__ xrec,
                                               float* __restrict__ resid,
                                               float* __restrict__ saeOut,
                                               float* __restrict__ accg) {
  __shared__ float sred[256];
  const int row = blockIdx.x, t = threadIdx.x;
  const float stdv = xstd[row];
  float loc = 0.f;
#pragma unroll
  for (int i = 0; i < 3; ++i) {
    size_t g = (size_t)row * ACT + t + i * 256;
    float n = xn[g], r = xrec[g];
    float res = n - r;
    resid[g] = res;
    saeOut[g] = r * stdv + n;
    loc += res * res;
  }
  float s = blockReduceSum256(loc, sred);
  if (t == 0) atomicAdd(&accg[0], s);
}

// ---------------- fused aux GEMMs (K=1024) + squared-error reductions ----------------
__global__ __launch_bounds__(256, 1) void k_gemm_aux(
    const __bf16* __restrict__ A1,  // acts_aux     [BATCH][NDEAD]
    const __bf16* __restrict__ A2,  // softmax dead [BATCH][NDEAD]
    const __bf16* __restrict__ Bt,  // [ACT][DICT], use first NDEAD cols
    const float* __restrict__ bdec,
    const float* __restrict__ resid,
    float* __restrict__ accg) {
  __shared__ __bf16 sA1[64 * 40];
  __shared__ __bf16 sA2[64 * 40];
  __shared__ __bf16 sB[128 * 40];
  __shared__ float sbias[128];
  __shared__ float sred[256];

  const int bn = blockIdx.x, bm = blockIdx.y;
  const int tid = threadIdx.x, lane = tid & 31, w = tid >> 5;
  const int wm = w >> 1, wn = w & 1;   // 4 x 2 wave grid, wave tile 16x64
  const int rowBase = bm * 64, colBase = bn * 128;
  if (tid < 128) sbias[tid] = bdec[colBase + tid];

  const v8f vz = {0.f,0.f,0.f,0.f,0.f,0.f,0.f,0.f};
  v8f acc1[4], acc2[4];
#pragma unroll
  for (int j = 0; j < 4; ++j) { acc1[j] = vz; acc2[j] = vz; }

  const int ra = tid >> 2, roff = (tid & 3) * 8;   // A tiles: 64x32
  const int ldr = tid >> 1, ldh = (tid & 1) * 16;  // B tile: 128x32
  const int mlo = lane & 15, hsel = lane >> 4;
  const int kb = 8 * hsel, kh = 16 * hsel;

  for (int k0 = 0; k0 < NDEAD; k0 += 32) {
    __syncthreads();
    uint4 a1 = *(const uint4*)(A1 + (size_t)(rowBase + ra) * NDEAD + k0 + roff);
    uint4 a2 = *(const uint4*)(A2 + (size_t)(rowBase + ra) * NDEAD + k0 + roff);
    const uint4* gb = (const uint4*)(Bt + (size_t)(colBase + ldr) * DICT + k0 + ldh);
    uint4 b0 = gb[0], b1 = gb[1];
    *(uint4*)&sA1[ra * 40 + roff] = a1;
    *(uint4*)&sA2[ra * 40 + roff] = a2;
    *(uint4*)&sB[ldr * 40 + ldh]     = b0;
    *(uint4*)&sB[ldr * 40 + ldh + 8] = b1;
    __syncthreads();

    FragBf a1F, a2F, bF[4];
    {
      int m = wm * 16 + mlo;
      a1F.q[0] = *(const uint4*)&sA1[m * 40 + kb];
      a1F.q[1] = *(const uint4*)&sA1[m * 40 + 16 + kb];
      a2F.q[0] = *(const uint4*)&sA2[m * 40 + kb];
      a2F.q[1] = *(const uint4*)&sA2[m * 40 + 16 + kb];
    }
#pragma unroll
    for (int j = 0; j < 4; ++j) {
      int n = wn * 64 + j * 16 + mlo;
      bF[j].q[0] = *(const uint4*)&sB[n * 40 + kb];
      bF[j].q[1] = *(const uint4*)&sB[n * 40 + 16 + kb];
    }
#pragma unroll
    for (int j = 0; j < 4; ++j) {
      acc1[j] = __builtin_amdgcn_wmma_f32_16x16x32_bf16(
          false, a1F.v, false, bF[j].v, (short)0, acc1[j], false, false);
      acc2[j] = __builtin_amdgcn_wmma_f32_16x16x32_bf16(
          false, a2F.v, false, bF[j].v, (short)0, acc2[j], false, false);
    }
  }
  __syncthreads();

  float sa = 0.f, ss = 0.f;
#pragma unroll
  for (int j = 0; j < 4; ++j)
#pragma unroll
    for (int r = 0; r < 8; ++r) {
      int row = rowBase + wm * 16 + 8 * hsel + r;
      int lcol = wn * 64 + j * 16 + mlo;
      float bias = sbias[lcol];
      float res = resid[(size_t)row * ACT + colBase + lcol];
      float da = (acc1[j][r] + bias) - res;
      float dv = (acc2[j][r] + bias) - res;
      sa += da * da;
      ss += dv * dv;
    }
  float ra_ = blockReduceSum256(sa, sred);
  if (tid == 0) atomicAdd(&accg[1], ra_);
  float rs_ = blockReduceSum256(ss, sred);
  if (tid == 0) atomicAdd(&accg[2], rs_);
}

// ---------------- finalize scalars ----------------
__global__ __launch_bounds__(256) void k_finalize(const float* __restrict__ accg,
                                                  const unsigned char* __restrict__ dead,
                                                  float* __restrict__ outScal) {
  __shared__ float sred[256];
  const int t = threadIdx.x;
  float c = 0.f;
  for (int i = t; i < DICT; i += 256) c += dead[i] ? 1.f : 0.f;
  float nd = blockReduceSum256(c, sred);
  if (t == 0) {
    const float invBA = 1.f / ((float)BATCH * (float)ACT);
    float l2      = accg[0] * invBA;
    float l2aux   = 0.03125f * accg[1] * invBA;
    float l2soft  = 0.01f    * accg[2] * invBA;
    float l1norm  = accg[3] / (float)BATCH;
    float l1loss  = 0.001f * l1norm;
    float l0norm  = accg[4] / (float)BATCH;
    outScal[0] = l2 + l1loss + l2aux + l2soft;  // loss
    outScal[1] = l2;
    outScal[2] = l1loss;
    outScal[3] = l0norm;
    outScal[4] = l1norm;
    outScal[5] = l2aux;
    outScal[6] = l2soft;
    outScal[7] = nd;                            // num_dead
  }
}

extern "C" void kernel_launch(void* const* d_in, const int* in_sizes, int n_in,
                              void* d_out, int out_size, void* d_ws, size_t ws_size,
                              hipStream_t stream) {
  (void)in_sizes; (void)n_in; (void)out_size; (void)ws_size;
  const float* x    = (const float*)d_in[0];
  const float* Wenc = (const float*)d_in[1];
  const float* Wdec = (const float*)d_in[2];
  const float* benc = (const float*)d_in[3];
  const float* bdec = (const float*)d_in[4];
  const float* q    = (const float*)d_in[5];
  const unsigned char* dead = (const unsigned char*)d_in[6];
  float* out = (float*)d_out;

  char* ws = (char*)d_ws;
  float*  xn      = (float*)(ws + OFF_XN);
  float*  xstd    = (float*)(ws + OFF_XSTD);
  __bf16* xcent   = (__bf16*)(ws + OFF_XCENT);
  __bf16* WencT   = (__bf16*)(ws + OFF_WENCT);
  __bf16* WdecT   = (__bf16*)(ws + OFF_WDECT);
  __bf16* thrBf   = (__bf16*)(ws + OFF_THRBF);
  float*  preDead = (float*)(ws + OFF_PREDEAD);
  float*  Mb      = (float*)(ws + OFF_MB);
  float*  Sb      = (float*)(ws + OFF_SB);
  float*  L1p     = (float*)(ws + OFF_L1P);
  float*  L0p     = (float*)(ws + OFF_L0P);
  __bf16* aux     = (__bf16*)(ws + OFF_AUX);
  __bf16* softd   = (__bf16*)(ws + OFF_SOFTD);
  float*  xrec    = (float*)(ws + OFF_XREC);
  float*  resid   = (float*)(ws + OFF_RESID);
  float*  accg    = (float*)(ws + OFF_ACC);

  k_init<<<1, 64, 0, stream>>>(accg);
  k_norm<<<BATCH, 256, 0, stream>>>(x, bdec, xn, xstd, xcent);
  k_transpose<<<dim3(DICT / 32, ACT / 32), 256, 0, stream>>>(Wenc, WencT, ACT, DICT);
  k_transpose<<<dim3(ACT / 32, DICT / 32), 256, 0, stream>>>(Wdec, WdecT, DICT, ACT);
  k_gemm_enc<<<dim3(DICT / 128, BATCH / 128), 256, 0, stream>>>(
      xcent, WencT, benc, q, out + OUT_ACTS, thrBf, preDead, Mb, Sb, L1p, L0p);
  k_rowaux<<<BATCH, 512, 0, stream>>>(preDead, Mb, Sb, L1p, L0p, softd, aux, accg);
  k_gemm_dec<<<dim3(ACT / 128, BATCH / 128), 256, 0, stream>>>(thrBf, WdecT, bdec, xrec);
  k_resid<<<BATCH, 256, 0, stream>>>(xn, xstd, xrec, resid, out + OUT_SAE, accg);
  k_gemm_aux<<<dim3(ACT / 128, BATCH / 64), 256, 0, stream>>>(
      aux, softd, WdecT, bdec, resid, accg);
  k_finalize<<<1, 256, 0, stream>>>(accg, dead, out + OUT_SCAL);
}
